// HeteroGraph_17325898072215
// MI455X (gfx1250) — matrix operand order repaired and checked by reading, not compiled, not run on hardware
//
#include <hip/hip_runtime.h>
#include <hip/hip_bf16.h>

// ---------------------------------------------------------------------------
// Hetero-GNN forward for gfx1250 (MI455X).
// Features stored bf16 (halves gather bandwidth), aggregation in fp32 atomics,
// all H=128 dense transforms via v_wmma_f32_16x16x32_bf16, weights staged to
// LDS via the Tensor Data Mover (tensor_load_to_lds + s_wait_tensorcnt).
// All row counts padded to multiples of 128 so the GEMM has no bounds checks.
// ---------------------------------------------------------------------------

constexpr int HD = 128;          // hidden channels
constexpr int NGR = 2048;        // num graphs

typedef __attribute__((ext_vector_type(16))) __bf16 v16bf;
typedef __attribute__((ext_vector_type(8)))  float  v8f;
typedef __attribute__((ext_vector_type(4)))  unsigned int u32x4;
typedef __attribute__((ext_vector_type(4)))  int i32x4;
typedef __attribute__((ext_vector_type(8)))  int i32x8;

#if __has_builtin(__builtin_amdgcn_tensor_load_to_lds) && __has_builtin(__builtin_amdgcn_s_wait_tensorcnt)
#define GNN_USE_TDM 1
#else
#define GNN_USE_TDM 0
#endif

__device__ __forceinline__ unsigned short f32_to_bf16(float f) {
  unsigned u = __float_as_uint(f);
  u += 0x7fffu + ((u >> 16) & 1u);            // round-to-nearest-even
  return (unsigned short)(u >> 16);
}
__device__ __forceinline__ float bf16_to_f32(unsigned short h) {
  return __uint_as_float(((unsigned)h) << 16);
}
__device__ __forceinline__ unsigned pack2_bf16(float a, float b) {
  return (unsigned)f32_to_bf16(a) | ((unsigned)f32_to_bf16(b) << 16);
}

// ---------------- projection: y = x @ W + b, stored bf16 -------------------
__global__ void proj_kernel(const float* __restrict__ x, const float* __restrict__ W,
                            const float* __restrict__ b, unsigned short* __restrict__ out,
                            int N, int in_dim) {
  long long tid = (long long)blockIdx.x * blockDim.x + threadIdx.x;
  if (tid >= (long long)N * HD) return;
  int n = (int)(tid >> 7);
  int j = (int)tid & 127;
  float acc = b[j];
  const float* xr = x + (long long)n * in_dim;
  for (int k = 0; k < in_dim; k++) acc += xr[k] * W[k * HD + j];
  out[tid] = f32_to_bf16(acc);
}

// ------- weight prep: sum up to 4 f32 128x128 mats, emit WMMA-frag layout ---
// out[(((s*2+half)*128)+col)*8+v] packs (W[k,col],W[k+1,col]), k=32s+8*half+2*(v&3)+16*(v>>2)
__global__ void wsum_cvt_kernel(const float* __restrict__ a, const float* __restrict__ b,
                                const float* __restrict__ c, const float* __restrict__ d,
                                unsigned int* __restrict__ out) {
  int t = blockIdx.x * blockDim.x + threadIdx.x;
  if (t >= 8192) return;
  int v = t & 7, col = (t >> 3) & 127, hf = (t >> 10) & 1, s = t >> 11;
  int k = 32 * s + 8 * hf + 2 * (v & 3) + 16 * (v >> 2);
  int i0 = k * HD + col, i1 = (k + 1) * HD + col;
  float x0 = a[i0], x1 = a[i1];
  if (b) { x0 += b[i0]; x1 += b[i1]; }
  if (c) { x0 += c[i0]; x1 += c[i1]; }
  if (d) { x0 += d[i0]; x1 += d[i1]; }
  out[t] = pack2_bf16(x0, x1);
}

__global__ void bsum_kernel(const float* __restrict__ a, const float* __restrict__ b,
                            const float* __restrict__ c, const float* __restrict__ d,
                            float* __restrict__ out) {
  int j = threadIdx.x;
  float v = a[j];
  if (b) v += b[j];
  if (c) v += c[j];
  if (d) v += d[j];
  out[j] = v;
}

// ---------------- edge scatter: agg[dst] += feat[src] (bf16 -> f32) --------
__global__ void scatter_kernel(const unsigned short* __restrict__ feat,
                               const int* __restrict__ srci, const int* __restrict__ dsti,
                               int E, float* __restrict__ agg) {
  long long tid = (long long)blockIdx.x * blockDim.x + threadIdx.x;
  int e = (int)(tid >> 5);
  if (e >= E) return;
  int c = ((int)tid & 31) << 2;     // 4 columns per lane
  long long s = srci[e], d = dsti[e];
  uint2 u = *(const uint2*)(feat + s * HD + c);
  float* q = agg + d * HD + c;
  unsafeAtomicAdd(q + 0, bf16_to_f32((unsigned short)(u.x & 0xffffu)));
  unsafeAtomicAdd(q + 1, bf16_to_f32((unsigned short)(u.x >> 16)));
  unsafeAtomicAdd(q + 2, bf16_to_f32((unsigned short)(u.y & 0xffffu)));
  unsafeAtomicAdd(q + 3, bf16_to_f32((unsigned short)(u.y >> 16)));
}

// ---------------- WMMA GEMM --------------------------------------------------
// INIT=true : C[M,128]  = A[M,128] @ W[128,128] + bias[col]   (plain store)
// INIT=false: C[M,128] += A[M,128] @ W[128,128]               (C preloaded as
//                                                              WMMA C operand)
// W pre-packed (wsum_cvt layout), staged in 32KB LDS via TDM; A either bf16
// or f32 (converted in-register). 8 waves/block, 16 rows/wave, Mp % 128 == 0.
union ABpack { uint4 u[2]; v16bf v; };

template <bool ABF, bool INIT>
__global__ __launch_bounds__(256) void gemm128_kernel(float* __restrict__ C,
                                                      const void* __restrict__ Aptr,
                                                      const unsigned int* __restrict__ wfrag,
                                                      const float* __restrict__ bias,
                                                      int Mp) {
  __shared__ __align__(16) unsigned int lds[8192];
#if GNN_USE_TDM
  if (threadIdx.x == 0) {
    unsigned long long ga = (unsigned long long)wfrag;
    unsigned ldsb = (unsigned)(unsigned long long)(void*)lds;
    u32x4 g0 = { 1u,                                   // count=1 valid descriptor
                 ldsb,                                 // lds_addr (bytes)
                 (unsigned)(ga & 0xffffffffu),         // global_addr[31:0]
                 (unsigned)((ga >> 32) & 0x01ffffffu) | (2u << 30) };  // addr[56:32] | type=2
    i32x8 g1 = { (int)(2u << 16),        // data_size = 4B
                 (int)(0x2000u << 16),   // tensor_dim0 = 8192 (bits 63:48)
                 (int)(1u << 16),        // tensor_dim1 = 1    (bits 95:80)
                 (int)(0x2000u << 16),   // tile_dim0   = 8192 (bits 127:112)
                 1,                      // tile_dim1   = 1
                 8192,                   // tensor_dim0_stride
                 0, 0 };
    i32x4 gz = { 0, 0, 0, 0 };
#if __clang_major__ >= 23
    i32x8 gz8 = { 0, 0, 0, 0, 0, 0, 0, 0 };
    __builtin_amdgcn_tensor_load_to_lds(g0, g1, gz, gz, gz8, 0);
#else
    __builtin_amdgcn_tensor_load_to_lds(g0, g1, gz, gz, 0);
#endif
    __builtin_amdgcn_s_wait_tensorcnt(0);
  }
#else
  {
    const uint4* gw = (const uint4*)wfrag;
    uint4* lw = (uint4*)lds;
    for (int i = threadIdx.x; i < 2048; i += 256) lw[i] = gw[i];
  }
#endif
  __syncthreads();

  int wave = threadIdx.x >> 5, lane = threadIdx.x & 31;
  int hf = lane >> 4, l16 = lane & 15;
  long long rowA = (long long)blockIdx.x * 128 + wave * 16 + l16;   // < Mp by construction

  // A fragments per ISA 16-bit A layout:
  // lanes 0-15: K {32s+0..7, 32s+16..23}; lanes 16-31: K {32s+8..15, 32s+24..31}
  uint4 a[4][2];
  if (ABF) {
    const unsigned short* A = (const unsigned short*)Aptr + rowA * HD;
#pragma unroll
    for (int s = 0; s < 4; s++) {
      int k0 = 32 * s + hf * 8;
      a[s][0] = *(const uint4*)(A + k0);
      a[s][1] = *(const uint4*)(A + k0 + 16);
    }
  } else {
    const float* A = (const float*)Aptr + rowA * HD;
#pragma unroll
    for (int s = 0; s < 4; s++) {
      int k0 = 32 * s + hf * 8;
      float4 f0 = *(const float4*)(A + k0);
      float4 f1 = *(const float4*)(A + k0 + 4);
      float4 f2 = *(const float4*)(A + k0 + 16);
      float4 f3 = *(const float4*)(A + k0 + 20);
      a[s][0] = make_uint4(pack2_bf16(f0.x, f0.y), pack2_bf16(f0.z, f0.w),
                           pack2_bf16(f1.x, f1.y), pack2_bf16(f1.z, f1.w));
      a[s][1] = make_uint4(pack2_bf16(f2.x, f2.y), pack2_bf16(f2.z, f2.w),
                           pack2_bf16(f3.x, f3.y), pack2_bf16(f3.z, f3.w));
    }
  }

  long long rb = (long long)blockIdx.x * 128 + wave * 16 + hf * 8;
#pragma unroll
  for (int n = 0; n < 8; n++) {
    // C/D layout: VGPR i -> row rb+i, col = 16n + l16
    v8f acc;
    if (INIT) {
      float bv = bias[n * 16 + l16];
#pragma unroll
      for (int i = 0; i < 8; i++) acc[i] = bv;
    } else {
#pragma unroll
      for (int i = 0; i < 8; i++) acc[i] = C[(rb + i) * HD + n * 16 + l16];
    }
#pragma unroll
    for (int s = 0; s < 4; s++) {
      int bidx = ((((s << 1) + hf) << 7) + (n << 4) + l16) << 3;
      ABpack ub;
      ub.u[0] = *(const uint4*)&lds[bidx];
      ub.u[1] = *(const uint4*)&lds[bidx + 4];
      ABpack ua;
      ua.u[0] = a[s][0];
      ua.u[1] = a[s][1];
      acc = __builtin_amdgcn_wmma_f32_16x16x32_bf16(
          /*neg_a=*/false, ua.v, /*neg_b=*/false, ub.v,
          /*c_mod=*/(short)0, acc, /*reuse_a=*/false, /*reuse_b=*/false);
    }
#pragma unroll
    for (int i = 0; i < 8; i++) C[(rb + i) * HD + n * 16 + l16] = acc[i];
  }
}

// ---------------- ReLU variants -------------------------------------------
__global__ void relu_bf16_kernel(const float* __restrict__ in, unsigned short* __restrict__ out,
                                 long long total) {
  long long tid = (long long)blockIdx.x * blockDim.x + threadIdx.x;
  if (tid >= total) return;
  out[tid] = f32_to_bf16(fmaxf(in[tid], 0.f));
}
__global__ void relu_f32_kernel(float* __restrict__ p, long long total) {
  long long tid = (long long)blockIdx.x * blockDim.x + threadIdx.x;
  if (tid >= total) return;
  p[tid] = fmaxf(p[tid], 0.f);
}

// ---------------- pooling + head ------------------------------------------
__global__ void count_kernel(const int* __restrict__ batch, int n, float* __restrict__ counts) {
  int tid = blockIdx.x * blockDim.x + threadIdx.x;
  if (tid >= n) return;
  unsafeAtomicAdd(&counts[batch[tid]], 1.0f);
}

__global__ void pool_kernel(const float* __restrict__ op, const int* __restrict__ batch,
                            int n, float* __restrict__ pooled) {
  long long tid = (long long)blockIdx.x * blockDim.x + threadIdx.x;
  int e = (int)(tid >> 5);
  if (e >= n) return;
  int c = ((int)tid & 31) << 2;
  int g = batch[e];
  float4 v = *(const float4*)(op + (long long)e * HD + c);
  float* q = pooled + (long long)g * HD + c;
  unsafeAtomicAdd(q + 0, v.x);
  unsafeAtomicAdd(q + 1, v.y);
  unsafeAtomicAdd(q + 2, v.z);
  unsafeAtomicAdd(q + 3, v.w);
}

__global__ void final_kernel(const float* __restrict__ pooled, const float* __restrict__ counts,
                             const float* __restrict__ linW, const float* __restrict__ linb,
                             float* __restrict__ out) {
  __shared__ float sm[128];
  int g = blockIdx.x, j = threadIdx.x;
  float cnt = fmaxf(counts[g], 1.0f);
  sm[j] = pooled[(long long)g * HD + j] / cnt * linW[j];
  __syncthreads();
  for (int s = 64; s > 0; s >>= 1) {
    if (j < s) sm[j] += sm[j + s];
    __syncthreads();
  }
  if (j == 0) out[g] = sm[0] + linb[0];
}

// ---------------------------------------------------------------------------
extern "C" void kernel_launch(void* const* d_in, const int* in_sizes, int n_in,
                              void* d_out, int out_size, void* d_ws, size_t ws_size,
                              hipStream_t stream) {
  (void)in_sizes; (void)n_in; (void)out_size; (void)ws_size;

  // type ids: 0=operator 1=table 2=column 3=predicate
  static const int NT[4]  = {150000, 50000, 300000, 150000};
  static const int NP[4]  = {150016, 50048, 300032, 150016};   // padded to %128
  static const int IND[4] = {4, 2, 16, 1};
  static const int ETS[7] = {1, 3, 2, 2, 0, 1, 2};   // src type per ETYPE
  static const int ETD[7] = {0, 0, 0, 3, 0, 1, 2};   // dst type per ETYPE
  static const int ETE[7] = {150000, 300000, 600000, 450000, 300000, 50000, 300000};
  // input indices (setup_inputs dict order, nested dicts in insertion order)
  const int I_PROJ = 4;            // 4 types x (W,b) -> 4..11
  const int I_CONV[2] = {12, 33};  // 7 x (Wrel,brel,Wroot) each
  const int I_LINW = 54, I_LINB = 55, I_EDGE = 56, I_BATCH = 63;

  char* ws = (char*)d_ws;
  auto alloc = [&](size_t bytes) -> char* {
    char* p = ws;
    ws += (bytes + 255) & ~(size_t)255;
    return p;
  };
  float* of[4];
  unsigned short* hb[4];
  for (int t = 0; t < 4; t++) of[t] = (float*)alloc((size_t)NP[t] * HD * 4);
  for (int t = 0; t < 4; t++) hb[t] = (unsigned short*)alloc((size_t)NP[t] * HD * 2);
  float* agg = (float*)alloc((size_t)NP[2] * HD * 4);
  unsigned int* wfrag = (unsigned int*)alloc((size_t)22 * 8192 * 4);
  float* brels = (float*)alloc((size_t)8 * HD * 4);
  float* pooled = (float*)alloc((size_t)NGR * HD * 4);
  float* counts = (float*)alloc((size_t)NGR * 4);

  // zero the pad rows of hb once per launch (deterministic)
  for (int t = 0; t < 4; t++)
    hipMemsetAsync(hb[t] + (size_t)NT[t] * HD, 0, (size_t)(NP[t] - NT[t]) * HD * 2, stream);

  // ---- weight prep: bf16 fragment-layout Wrel, plus folded Wroot/brel sums ----
  for (int c = 0; c < 2; c++) {
    int pb = I_CONV[c];
    const float *Wr[7], *Br[7];
    for (int et = 0; et < 7; et++) {
      const float* wrel = (const float*)d_in[pb + et * 3 + 0];
      Br[et] = (const float*)d_in[pb + et * 3 + 1];
      Wr[et] = (const float*)d_in[pb + et * 3 + 2];
      wsum_cvt_kernel<<<32, 256, 0, stream>>>(wrel, nullptr, nullptr, nullptr,
                                              wfrag + (size_t)(c * 11 + et) * 8192);
    }
    // root sums per dst type: op={et 0,1,2,4}, tb={5}, col={6}, pr={3}
    wsum_cvt_kernel<<<32, 256, 0, stream>>>(Wr[0], Wr[1], Wr[2], Wr[4], wfrag + (size_t)(c * 11 + 7 + 0) * 8192);
    wsum_cvt_kernel<<<32, 256, 0, stream>>>(Wr[5], nullptr, nullptr, nullptr, wfrag + (size_t)(c * 11 + 7 + 1) * 8192);
    wsum_cvt_kernel<<<32, 256, 0, stream>>>(Wr[6], nullptr, nullptr, nullptr, wfrag + (size_t)(c * 11 + 7 + 2) * 8192);
    wsum_cvt_kernel<<<32, 256, 0, stream>>>(Wr[3], nullptr, nullptr, nullptr, wfrag + (size_t)(c * 11 + 7 + 3) * 8192);
    bsum_kernel<<<1, 128, 0, stream>>>(Br[0], Br[1], Br[2], Br[4], brels + (c * 4 + 0) * HD);
    bsum_kernel<<<1, 128, 0, stream>>>(Br[5], nullptr, nullptr, nullptr, brels + (c * 4 + 1) * HD);
    bsum_kernel<<<1, 128, 0, stream>>>(Br[6], nullptr, nullptr, nullptr, brels + (c * 4 + 2) * HD);
    bsum_kernel<<<1, 128, 0, stream>>>(Br[3], nullptr, nullptr, nullptr, brels + (c * 4 + 3) * HD);
  }

  // ---- projections -> bf16 features ----
  for (int t = 0; t < 4; t++) {
    long long tot = (long long)NT[t] * HD;
    proj_kernel<<<(unsigned)((tot + 255) / 256), 256, 0, stream>>>(
        (const float*)d_in[t], (const float*)d_in[I_PROJ + 2 * t],
        (const float*)d_in[I_PROJ + 2 * t + 1], hb[t], NT[t], IND[t]);
  }

  auto run_conv = [&](int c, bool only_op) {
    // root GEMM first: of[d] = hb[d] @ Wroot_sum + brel_sum (writes all pad rows)
    for (int t = 0; t < 4; t++) {
      if (only_op && t != 0) continue;
      gemm128_kernel<true, true><<<NP[t] / 128, 256, 0, stream>>>(
          of[t], hb[t], wfrag + (size_t)(c * 11 + 7 + t) * 8192, brels + (c * 4 + t) * HD, NP[t]);
    }
    // per edge type: zero agg (padded), scatter, of[d] += agg @ Wrel
    for (int et = 0; et < 7; et++) {
      int d = ETD[et];
      if (only_op && d != 0) continue;
      int s = ETS[et], E = ETE[et];
      hipMemsetAsync(agg, 0, (size_t)NP[d] * HD * 4, stream);
      const int* ei = (const int*)d_in[I_EDGE + et];  // row0=src, row1=dst
      long long tot = (long long)E * 32;
      scatter_kernel<<<(unsigned)((tot + 255) / 256), 256, 0, stream>>>(hb[s], ei, ei + E, E, agg);
      gemm128_kernel<false, false><<<NP[d] / 128, 256, 0, stream>>>(
          of[d], agg, wfrag + (size_t)(c * 11 + et) * 8192, nullptr, NP[d]);
    }
  };

  run_conv(0, /*only_op=*/false);
  for (int t = 0; t < 4; t++) {
    long long tot = (long long)NP[t] * HD;   // padded rows too -> hb pad stays finite
    relu_bf16_kernel<<<(unsigned)((tot + 255) / 256), 256, 0, stream>>>(of[t], hb[t], tot);
  }
  run_conv(1, /*only_op=*/true);  // only operator outputs are used downstream
  {
    long long tot = (long long)NT[0] * HD;
    relu_f32_kernel<<<(unsigned)((tot + 255) / 256), 256, 0, stream>>>(of[0], tot);
  }

  hipMemsetAsync(pooled, 0, (size_t)NGR * HD * 4, stream);
  hipMemsetAsync(counts, 0, (size_t)NGR * 4, stream);
  count_kernel<<<(150000 + 255) / 256, 256, 0, stream>>>((const int*)d_in[I_BATCH], 150000, counts);
  {
    long long tot = (long long)150000 * 32;
    pool_kernel<<<(unsigned)((tot + 255) / 256), 256, 0, stream>>>(
        of[0], (const int*)d_in[I_BATCH], 150000, pooled);
  }
  final_kernel<<<NGR, 128, 0, stream>>>(pooled, counts, (const float*)d_in[I_LINW],
                                        (const float*)d_in[I_LINB], (float*)d_out);
}
